// ImplicitNetworkGrid_79431125172520
// MI455X (gfx1250) — matrix operand order repaired and checked by reading, not compiled, not run on hardware
//
#include <hip/hip_runtime.h>

typedef __attribute__((ext_vector_type(16))) _Float16 v16h;
typedef __attribute__((ext_vector_type(8)))  float    v8f;

#define NLEV   16
#define TSIZE  (1u << 19)
#define HID    256
#define D0PAD  128            // 121 padded to 128 (K of layer 0)
#define ISTR   136            // inp / W0 row stride in halfs (128+8 -> conflict-free b128)
#define ASTR   264            // act / W1 row stride in halfs (256+8)
#define W0_HALFS (HID * ISTR) // 34816  (69632 B)
#define W1_HALFS (HID * ASTR) // 67584  (135168 B)
#define NPTS   256            // points per block (8 waves x 2 M-tiles x 16 rows)

// ceil(16 * (2^(7/15))^l) for l=0..15
__constant__ int c_res[NLEV] = {16,23,31,43,59,81,112,155,213,295,407,562,777,1073,1483,2048};

__device__ __forceinline__ float softplus100(float v) {
  float t = v * 100.0f;
  if (t > 20.0f) return v;
  return __logf(1.0f + __expf(t)) * 0.01f;
}

union H4      { _Float16 h[4];  uint2 u; };
union F16Frag { uint4    q[2];  v16h  v; };

// CDNA5 async global->LDS copy (ASYNCcnt); 16B per lane per instruction.
__device__ __forceinline__ void async_copy16(unsigned lds_addr, const void* gsrc) {
  asm volatile("global_load_async_to_lds_b128 %0, %1, off"
               :: "v"(lds_addr), "v"(gsrc) : "memory");
}
__device__ __forceinline__ void wait_async0() {
  asm volatile("s_wait_asynccnt 0x0" ::: "memory");
}

// Convert weights to f16 in the exact padded LDS layouts (so staging is a flat copy).
extern "C" __global__ __launch_bounds__(256)
void prep_weights(const float* __restrict__ W0, const float* __restrict__ W1,
                  _Float16* __restrict__ w0p, _Float16* __restrict__ w1p)
{
  int i = blockIdx.x * blockDim.x + threadIdx.x;
  if (i < W0_HALFS) {
    int j = i / ISTR, c = i % ISTR;
    float v = 0.0f;
    if (c < 64)       v = W0[j * 121 + 57 + c];    // hash-feature cols
    else if (c < 121) v = W0[j * 121 + (c - 64)];  // x + sin/cos embed cols
    w0p[i] = (_Float16)v;
  } else if (i < W0_HALFS + W1_HALFS) {
    int k = i - W0_HALFS;
    int j = k / ASTR, c = k % ASTR;
    float v = (c < HID) ? W1[j * HID + c] : 0.0f;
    w1p[k] = (_Float16)v;
  }
}

extern "C" __global__ __launch_bounds__(256, 2)
void sdf_fused(const float* __restrict__ x, const float* __restrict__ table,
               const _Float16* __restrict__ w0p, const _Float16* __restrict__ w1p,
               const float* __restrict__ b0, const float* __restrict__ b1,
               const float* __restrict__ W2, const float* __restrict__ b2,
               float* __restrict__ out)
{
  // Phase-overlapped LDS regions (270336 B of the 320 KB WGP LDS):
  __shared__ __align__(16) _Float16 sR1[W1_HALFS];       // phase1: sInp[256][ISTR]; phase2: W1[256][ASTR]
  __shared__ __align__(16) _Float16 sR2[NPTS * ASTR];    // phase1: W0[256][ISTR];  phase2: sAct[256][ASTR]
  _Float16* const sInp = sR1;
  _Float16* const sW1  = sR1;
  _Float16* const sW0  = sR2;
  _Float16* const sAct = sR2;

  const int tid   = threadIdx.x;
  const int gbase = blockIdx.x * NPTS;

  // ---- kick off async staging of W0 into LDS (overlaps the encode phase) ----
  {
    const unsigned ldsW0 = (unsigned)(size_t)(void*)sW0;
    const char* g = (const char*)w0p;
    #pragma unroll
    for (int i = 0; i < 17; i++) {                  // 17*256*16B = 69632 B
      int c = tid + i * 256;
      async_copy16(ldsW0 + c * 16, g + c * 16);
    }
  }

  const float px = x[(size_t)(gbase + tid) * 3 + 0];
  const float py = x[(size_t)(gbase + tid) * 3 + 1];
  const float pz = x[(size_t)(gbase + tid) * 3 + 2];
  _Float16* row = sInp + tid * ISTR;

  // ---------------- encode phase (one thread per point) ----------------
  row[64] = (_Float16)px; row[65] = (_Float16)py; row[66] = (_Float16)pz;
  {
    float fx = px, fy = py, fz = pz;
    #pragma unroll
    for (int m = 0; m < 9; m++) {
      row[67 + m * 3 + 0] = (_Float16)__sinf(fx);
      row[67 + m * 3 + 1] = (_Float16)__sinf(fy);
      row[67 + m * 3 + 2] = (_Float16)__sinf(fz);
      row[94 + m * 3 + 0] = (_Float16)__cosf(fx);
      row[94 + m * 3 + 1] = (_Float16)__cosf(fy);
      row[94 + m * 3 + 2] = (_Float16)__cosf(fz);
      fx *= 2.0f; fy *= 2.0f; fz *= 2.0f;
    }
  }
  #pragma unroll
  for (int c = 121; c < 128; c++) row[c] = (_Float16)0.0f;

  #pragma unroll 2
  for (int l = 0; l < NLEV; l++) {
    const int res = c_res[l];
    const float gx = px * res, gy = py * res, gz = pz * res;
    const float flx = floorf(gx), fly = floorf(gy), flz = floorf(gz);
    const float frx = gx - flx, fry = gy - fly, frz = gz - flz;
    const int ix = min(max((int)flx, 0), res - 1);
    const int iy = min(max((int)fly, 0), res - 1);
    const int iz = min(max((int)flz, 0), res - 1);
    const float* tb = table + (size_t)l * TSIZE * 4;
    const bool dense = (res <= 79);                // (res+1)^3 <= 2^19
    const int r1 = res + 1;
    float a0 = 0.f, a1 = 0.f, a2 = 0.f, a3 = 0.f;
    #pragma unroll
    for (int c = 0; c < 8; c++) {
      const int cx = ix + (c & 1), cy = iy + ((c >> 1) & 1), cz = iz + ((c >> 2) & 1);
      unsigned idx;
      if (dense) idx = (unsigned)(cx + cy * r1 + cz * r1 * r1);
      else       idx = ((unsigned)cx ^ ((unsigned)cy * 2654435761u)
                                     ^ ((unsigned)cz * 805459861u)) & (TSIZE - 1u);
      const float4 f = *(const float4*)(tb + (size_t)idx * 4);
      const float w = ((c & 1) ? frx : 1.0f - frx) *
                      ((c & 2) ? fry : 1.0f - fry) *
                      ((c & 4) ? frz : 1.0f - frz);
      a0 += w * f.x; a1 += w * f.y; a2 += w * f.z; a3 += w * f.w;
    }
    H4 t;
    t.h[0] = (_Float16)a0; t.h[1] = (_Float16)a1;
    t.h[2] = (_Float16)a2; t.h[3] = (_Float16)a3;
    *(uint2*)(row + l * 4) = t.u;
  }

  wait_async0();       // W0 staged
  __syncthreads();     // sInp + W0 visible to all waves

  // ---------------- GEMM phase: 2 M-tiles per wave ----------------
  const int lane = tid & 31;
  const int wv   = tid >> 5;          // wave 0..7 owns rows wv*32..wv*32+31
  const int ln15 = lane & 15;
  const int hsel = lane >> 4;
  const int m0   = wv * 32;           // tile 0 row base
  const int ar0  = m0 + ln15;         // A rows for this lane
  const int ar1  = m0 + 16 + ln15;

  v8f acc0[16], acc1[16];
  const v8f vzero = {0.f,0.f,0.f,0.f,0.f,0.f,0.f,0.f};
  #pragma unroll
  for (int nt = 0; nt < 16; nt++) { acc0[nt] = vzero; acc1[nt] = vzero; }

  // Layer 0: [32 x 128] @ [128 x 256]; B double-buffered, each B feeds two WMMAs
  for (int kc = 0; kc < 4; kc++) {
    F16Frag A0, A1;
    const _Float16* ap0 = sInp + ar0 * ISTR + kc * 32;
    const _Float16* ap1 = sInp + ar1 * ISTR + kc * 32;
    A0.q[0] = *(const uint4*)(ap0 + hsel * 8);
    A0.q[1] = *(const uint4*)(ap0 + 16 + hsel * 8);
    A1.q[0] = *(const uint4*)(ap1 + hsel * 8);
    A1.q[1] = *(const uint4*)(ap1 + 16 + hsel * 8);

    const _Float16* bbase = sW0 + ln15 * ISTR + kc * 32 + hsel * 8;
    F16Frag Bcur, Bnxt;
    Bcur.q[0] = *(const uint4*)(bbase);
    Bcur.q[1] = *(const uint4*)(bbase + 16);
    #pragma unroll
    for (int nt = 0; nt < 16; nt++) {
      if (nt < 15) {                                  // prefetch next B fragment
        const _Float16* bp = bbase + (nt + 1) * 16 * ISTR;
        Bnxt.q[0] = *(const uint4*)(bp);
        Bnxt.q[1] = *(const uint4*)(bp + 16);
      }
      acc0[nt] = __builtin_amdgcn_wmma_f32_16x16x32_f16(
          false, A0.v, false, Bcur.v, (short)0, acc0[nt], false, false);
      acc1[nt] = __builtin_amdgcn_wmma_f32_16x16x32_f16(
          false, A1.v, false, Bcur.v, (short)0, acc1[nt], false, false);
      Bcur = Bnxt;
    }
  }

  __syncthreads();     // everyone done reading sInp (R1) and W0 (R2)

  // ---- async-stage W1 into R1 while we run the softplus epilogue ----
  {
    const unsigned ldsW1 = (unsigned)(size_t)(void*)sW1;
    const char* g = (const char*)w1p;
    #pragma unroll
    for (int i = 0; i < 33; i++) {                 // 33*256*16B = 135168 B
      int c = tid + i * 256;
      async_copy16(ldsW1 + c * 16, g + c * 16);
    }
  }

  // epilogue 0: bias + softplus100 -> f16 activations (wave-local rows of sAct)
  #pragma unroll
  for (int nt = 0; nt < 16; nt++) {
    const int col = nt * 16 + ln15;
    const float bias = b0[col];
    #pragma unroll
    for (int r = 0; r < 8; r++) {
      const float v0 = softplus100(acc0[nt][r] + bias);
      const float v1 = softplus100(acc1[nt][r] + bias);
      sAct[(m0 + hsel * 8 + r) * ASTR + col]      = (_Float16)v0;
      sAct[(m0 + 16 + hsel * 8 + r) * ASTR + col] = (_Float16)v1;
    }
  }

  #pragma unroll
  for (int nt = 0; nt < 16; nt++) { acc0[nt] = vzero; acc1[nt] = vzero; }

  wait_async0();       // W1 staged
  __syncthreads();     // W1 visible (sAct rows are wave-local anyway)

  // Layer 1: [32 x 256] @ [256 x 256]; B double-buffered
  for (int kc = 0; kc < 8; kc++) {
    F16Frag A0, A1;
    const _Float16* ap0 = sAct + ar0 * ASTR + kc * 32;
    const _Float16* ap1 = sAct + ar1 * ASTR + kc * 32;
    A0.q[0] = *(const uint4*)(ap0 + hsel * 8);
    A0.q[1] = *(const uint4*)(ap0 + 16 + hsel * 8);
    A1.q[0] = *(const uint4*)(ap1 + hsel * 8);
    A1.q[1] = *(const uint4*)(ap1 + 16 + hsel * 8);

    const _Float16* bbase = sW1 + ln15 * ASTR + kc * 32 + hsel * 8;
    F16Frag Bcur, Bnxt;
    Bcur.q[0] = *(const uint4*)(bbase);
    Bcur.q[1] = *(const uint4*)(bbase + 16);
    #pragma unroll
    for (int nt = 0; nt < 16; nt++) {
      if (nt < 15) {
        const _Float16* bp = bbase + (nt + 1) * 16 * ASTR;
        Bnxt.q[0] = *(const uint4*)(bp);
        Bnxt.q[1] = *(const uint4*)(bp + 16);
      }
      acc0[nt] = __builtin_amdgcn_wmma_f32_16x16x32_f16(
          false, A0.v, false, Bcur.v, (short)0, acc0[nt], false, false);
      acc1[nt] = __builtin_amdgcn_wmma_f32_16x16x32_f16(
          false, A1.v, false, Bcur.v, (short)0, acc1[nt], false, false);
      Bcur = Bnxt;
    }
  }

  // epilogue 1 + output layer (both tiles)
  float part0[8], part1[8];
  #pragma unroll
  for (int r = 0; r < 8; r++) { part0[r] = 0.0f; part1[r] = 0.0f; }
  #pragma unroll
  for (int nt = 0; nt < 16; nt++) {
    const int col = nt * 16 + ln15;
    const float bias = b1[col];
    const float w2v = W2[col];
    #pragma unroll
    for (int r = 0; r < 8; r++) {
      part0[r] += softplus100(acc0[nt][r] + bias) * w2v;
      part1[r] += softplus100(acc1[nt][r] + bias) * w2v;
    }
  }
  const float bout = b2[0];
  #pragma unroll
  for (int r = 0; r < 8; r++) {
    float s0 = part0[r], s1 = part1[r];
    s0 += __shfl_xor(s0, 1, 32); s1 += __shfl_xor(s1, 1, 32);
    s0 += __shfl_xor(s0, 2, 32); s1 += __shfl_xor(s1, 2, 32);
    s0 += __shfl_xor(s0, 4, 32); s1 += __shfl_xor(s1, 4, 32);
    s0 += __shfl_xor(s0, 8, 32); s1 += __shfl_xor(s1, 8, 32);
    if (ln15 == 0) {
      out[gbase + m0 + hsel * 8 + r]      = s0 + bout;
      out[gbase + m0 + 16 + hsel * 8 + r] = s1 + bout;
    }
  }
}

extern "C" void kernel_launch(void* const* d_in, const int* in_sizes, int n_in,
                              void* d_out, int out_size, void* d_ws, size_t ws_size,
                              hipStream_t stream) {
  const float* x  = (const float*)d_in[0];
  const float* tb = (const float*)d_in[1];
  const float* W0 = (const float*)d_in[2];
  const float* b0 = (const float*)d_in[3];
  const float* W1 = (const float*)d_in[4];
  const float* b1 = (const float*)d_in[5];
  const float* W2 = (const float*)d_in[6];
  const float* b2 = (const float*)d_in[7];
  float* out = (float*)d_out;

  _Float16* w0p = (_Float16*)d_ws;          // padded [256][136]
  _Float16* w1p = w0p + W0_HALFS;           // padded [256][264]

  const int prep_elems = W0_HALFS + W1_HALFS;
  prep_weights<<<(prep_elems + 255) / 256, 256, 0, stream>>>(W0, W1, w0p, w1p);

  const int N = in_sizes[0] / 3;            // 524288
  sdf_fused<<<N / NPTS, 256, 0, stream>>>(x, tb, w0p, w1p, b0, b1, W2, b2, out);
}
